// DFINETransformer_90537910600294
// MI455X (gfx1250) — compile-verified, matrix operand order checked by reading, not run on hardware
//
#include <hip/hip_runtime.h>
#include <hip/hip_bf16.h>
#include <math.h>

// ---------------------------------------------------------------------------
// DFINE transformer decoder layer for gfx1250 (MI455X), wave32 + WMMA.
// GEMMs: v_wmma_f32_16x16x32_f16, 128x64 tiles, double-buffered LDS,
//        8 WMMAs per wave per K-step. Flash attention with DPP16 softmax
//        reductions (no ds_bpermute round-trips).
// ---------------------------------------------------------------------------

typedef __attribute__((ext_vector_type(16))) _Float16 v16h;
typedef __attribute__((ext_vector_type(8)))  float    v8f;

#define LDH 40   // LDS row stride in halfs (80B: 16B-aligned chunks, bank-skewed)

// Fragment loader matching CDNA5 16-bit A/B layout for 16x16x32 WMMA:
// lanes 0-15: row = lane, K chunks [0..7] and [16..23]
// lanes 16-31: row = lane-16, K chunks [8..15] and [24..31]
__device__ __forceinline__ v16h frag_ld(const _Float16* base, int lane) {
  const int lo = lane & 15, hi = lane >> 4;
  const _Float16* r = base + lo * LDH + hi * 8;
  v16h f;
#pragma unroll
  for (int i = 0; i < 8; ++i) f[i] = r[i];
#pragma unroll
  for (int i = 0; i < 8; ++i) f[8 + i] = r[16 + i];
  return f;
}

__device__ __forceinline__ v8f wmma32(v16h a, v16h b, v8f c) {
  return __builtin_amdgcn_wmma_f32_16x16x32_f16(false, a, false, b, (short)0, c,
                                                false, false);
}

// ---- DPP16 butterfly reduction within each 16-lane half-wave (VALU only) ----
template <int CTRL>
__device__ __forceinline__ float dpp_f(float x) {
  return __builtin_bit_cast(float,
      __builtin_amdgcn_update_dpp(0, __builtin_bit_cast(int, x), CTRL, 0xF, 0xF, true));
}
__device__ __forceinline__ float row16_max(float v) {
  v = fmaxf(v, dpp_f<0xB1>(v));    // quad_perm(1,0,3,2)  : xor 1
  v = fmaxf(v, dpp_f<0x4E>(v));    // quad_perm(2,3,0,1)  : xor 2
  v = fmaxf(v, dpp_f<0x141>(v));   // row_half_mirror     : xor 4 (groups uniform)
  v = fmaxf(v, dpp_f<0x140>(v));   // row_mirror          : xor 8
  return v;
}
__device__ __forceinline__ float row16_sum(float v) {
  v += dpp_f<0xB1>(v);
  v += dpp_f<0x4E>(v);
  v += dpp_f<0x141>(v);
  v += dpp_f<0x140>(v);
  return v;
}

// ---------------------------------------------------------------------------
// Generic GEMM: C[M,N] = act(A[M,K] @ W[N,K]^T + bias[N])
// Block = 128 threads (4 waves), tile 128x64, K-step 32, double-buffered LDS.
// Each wave owns 32 rows: 2 A-frags x 4 B-frags = 8 WMMAs per K-step.
// Requires M % 128 == 0, N % 64 == 0, K % 32 == 0 (true for all call sites).
// ---------------------------------------------------------------------------
enum { ACT_NONE = 0, ACT_RELU = 1, ACT_SIG = 2 };

template <int ACT>
__global__ void __launch_bounds__(128)
gemm_wmma(const float* __restrict__ A, const float* __restrict__ W,
          const float* __restrict__ bias, float* __restrict__ C,
          int M, int N, int K) {
  __shared__ _Float16 As[2][128 * LDH];   // 2 x 10240 B
  __shared__ _Float16 Bs[2][64 * LDH];    // 2 x  5120 B
  const int m0 = blockIdx.y * 128;
  const int n0 = blockIdx.x * 64;
  const int wave = threadIdx.x >> 5;
  const int lane = threadIdx.x & 31;

  v8f acc[2][4] = {};

  auto load_tile = [&](int k0, int buf) {
#pragma unroll
    for (int i = 0; i < 8; ++i) {           // A: 128x32 f32 = 1024 float4
      const int id  = threadIdx.x + i * 128;
      const int row = id >> 3;
      const int c4  = (id & 7) * 4;
      const float4 av = *(const float4*)(A + (size_t)(m0 + row) * K + k0 + c4);
      _Float16* ar = As[buf] + row * LDH + c4;
      ar[0] = (_Float16)av.x; ar[1] = (_Float16)av.y;
      ar[2] = (_Float16)av.z; ar[3] = (_Float16)av.w;
    }
#pragma unroll
    for (int i = 0; i < 4; ++i) {           // B: 64x32 f32 = 512 float4
      const int id  = threadIdx.x + i * 128;
      const int row = id >> 3;
      const int c4  = (id & 7) * 4;
      const float4 wv = *(const float4*)(W + (size_t)(n0 + row) * K + k0 + c4);
      _Float16* br = Bs[buf] + row * LDH + c4;
      br[0] = (_Float16)wv.x; br[1] = (_Float16)wv.y;
      br[2] = (_Float16)wv.z; br[3] = (_Float16)wv.w;
    }
  };

  load_tile(0, 0);
  __syncthreads();
  int buf = 0;
  for (int k0 = 0; k0 < K; k0 += 32) {
    if (k0 + 32 < K) {
      load_tile(k0 + 32, buf ^ 1);
      __builtin_prefetch(A + (size_t)(m0 + threadIdx.x) * K + k0 + 64, 0, 1);
      __builtin_prefetch(W + (size_t)(n0 + (threadIdx.x >> 1)) * K + k0 + 64, 0, 1);
    }
    const v16h a0 = frag_ld(As[buf] + (wave * 32 +  0) * LDH, lane);
    const v16h a1 = frag_ld(As[buf] + (wave * 32 + 16) * LDH, lane);
#pragma unroll
    for (int j = 0; j < 4; ++j) {
      const v16h b = frag_ld(Bs[buf] + j * 16 * LDH, lane);
      acc[0][j] = wmma32(a0, b, acc[0][j]);
      acc[1][j] = wmma32(a1, b, acc[1][j]);
    }
    __syncthreads();
    buf ^= 1;
  }

  const int lo = lane & 15, hi = lane >> 4;
#pragma unroll
  for (int r = 0; r < 2; ++r) {
#pragma unroll
    for (int j = 0; j < 4; ++j) {
      const int n = n0 + j * 16 + lo;
      const float bvv = bias[n];
#pragma unroll
      for (int i = 0; i < 8; ++i) {
        const int m = m0 + wave * 32 + r * 16 + hi * 8 + i;
        float v = acc[r][j][i] + bvv;
        if (ACT == ACT_RELU) v = fmaxf(v, 0.f);
        if (ACT == ACT_SIG)  v = 1.f / (1.f + __expf(-v));
        C[(size_t)m * N + n] = v;
      }
    }
  }
}

// ---------------------------------------------------------------------------
// Flash self-attention, one block = (b, h, 64-query tile), 4 waves.
// Q/K/V layout: [B, 300, NH*DH] fp32; head h at cols [h*32, h*32+32).
// ---------------------------------------------------------------------------
__global__ void __launch_bounds__(128)
attn_flash(const float* __restrict__ Q, const float* __restrict__ Kg,
           const float* __restrict__ Vg, float* __restrict__ O) {
  __shared__ _Float16 Qs[64 * LDH];
  __shared__ _Float16 Ks[32 * LDH];
  __shared__ _Float16 Vs[32 * LDH];       // transposed: [dh][key]
  __shared__ _Float16 Ps[4][16 * LDH];    // per-wave P staging
  const int qt = blockIdx.x, h = blockIdx.y, b = blockIdx.z;
  const int q0 = qt * 64;
  const int wave = threadIdx.x >> 5;
  const int lane = threadIdx.x & 31;
  const int lo = lane & 15, hi = lane >> 4;

  // Q tile, pre-scaled by 1/sqrt(DH)
#pragma unroll
  for (int i = 0; i < 4; ++i) {
    const int id = threadIdx.x + i * 128;
    const int row = id >> 3;
    const int c4 = (id & 7) * 4;
    int q = q0 + row; if (q > 299) q = 299;
    const float4 v = *(const float4*)(Q + ((size_t)b * 300 + q) * 256 + h * 32 + c4);
    const float s = 0.17677669529663687f;
    _Float16* r = Qs + row * LDH + c4;
    r[0] = (_Float16)(v.x * s); r[1] = (_Float16)(v.y * s);
    r[2] = (_Float16)(v.z * s); r[3] = (_Float16)(v.w * s);
  }
  __syncthreads();
  const v16h aq = frag_ld(Qs + wave * 16 * LDH, lane);

  float mrow[8], lrow[8];
#pragma unroll
  for (int i = 0; i < 8; ++i) { mrow[i] = -1e30f; lrow[i] = 0.f; }
  v8f o0 = {}; v8f o1 = {};

  for (int kb = 0; kb < 300; kb += 32) {
    __syncthreads();
#pragma unroll
    for (int i = 0; i < 2; ++i) {
      const int id = threadIdx.x + i * 128;
      const int row = id >> 3;
      const int c4 = (id & 7) * 4;
      const int key = kb + row;
      float4 kv = {}; float4 vv = {};
      if (key < 300) {
        kv = *(const float4*)(Kg + ((size_t)b * 300 + key) * 256 + h * 32 + c4);
        vv = *(const float4*)(Vg + ((size_t)b * 300 + key) * 256 + h * 32 + c4);
      }
      _Float16* kr = Ks + row * LDH + c4;
      kr[0] = (_Float16)kv.x; kr[1] = (_Float16)kv.y;
      kr[2] = (_Float16)kv.z; kr[3] = (_Float16)kv.w;
      Vs[(c4 + 0) * LDH + row] = (_Float16)vv.x;
      Vs[(c4 + 1) * LDH + row] = (_Float16)vv.y;
      Vs[(c4 + 2) * LDH + row] = (_Float16)vv.z;
      Vs[(c4 + 3) * LDH + row] = (_Float16)vv.w;
    }
    __syncthreads();

    const v16h bk0 = frag_ld(Ks, lane);
    const v16h bk1 = frag_ld(Ks + 16 * LDH, lane);
    v8f s0 = {}; v8f s1 = {};
    s0 = wmma32(aq, bk0, s0);
    s1 = wmma32(aq, bk1, s1);
    const v16h bv0 = frag_ld(Vs, lane);
    const v16h bv1 = frag_ld(Vs + 16 * LDH, lane);

    const bool msk0 = (kb + lo) >= 300;
    const bool msk1 = (kb + 16 + lo) >= 300;
    _Float16* prow = &Ps[wave][0];
#pragma unroll
    for (int i = 0; i < 8; ++i) {
      const float x0 = msk0 ? -1e30f : s0[i];
      const float x1 = msk1 ? -1e30f : s1[i];
      const float mx = row16_max(fmaxf(x0, x1));       // DPP16 butterfly
      const float mn = fmaxf(mrow[i], mx);
      const float sc = __expf(mrow[i] - mn);
      mrow[i] = mn;
      const float p0 = __expf(x0 - mn);
      const float p1 = __expf(x1 - mn);
      const float rs = row16_sum(p0 + p1);             // DPP16 butterfly
      lrow[i] = lrow[i] * sc + rs;
      o0[i] *= sc; o1[i] *= sc;
      prow[(i + 8 * hi) * LDH + lo]      = (_Float16)p0;
      prow[(i + 8 * hi) * LDH + 16 + lo] = (_Float16)p1;
    }
    __syncthreads();  // cross-lane LDS round-trip for P re-fragmentation
    const v16h ap = frag_ld(prow, lane);
    o0 = wmma32(ap, bv0, o0);
    o1 = wmma32(ap, bv1, o1);
  }

#pragma unroll
  for (int i = 0; i < 8; ++i) {
    const int q = q0 + wave * 16 + hi * 8 + i;
    if (q < 300) {
      const float inv = 1.f / fmaxf(lrow[i], 1e-20f);
      float* op = O + ((size_t)b * 300 + q) * 256 + h * 32;
      op[lo]      = o0[i] * inv;
      op[16 + lo] = o1[i] * inv;
    }
  }
}

// ---------------------------------------------------------------------------
// Elementwise / normalization kernels
// ---------------------------------------------------------------------------
__global__ void add2(const float* __restrict__ a, const float* __restrict__ b,
                     float* __restrict__ out, int n) {
  const int i = blockIdx.x * 256 + threadIdx.x;
  if (i < n) out[i] = a[i] + b[i];
}

__global__ void concat2(const float* __restrict__ a, const float* __restrict__ b,
                        float* __restrict__ out, int n) {
  const int i = blockIdx.x * 256 + threadIdx.x;
  if (i >= n) return;
  const int row = i >> 9, col = i & 511;
  out[i] = (col < 256) ? a[(size_t)row * 256 + col]
                       : b[(size_t)row * 256 + (col - 256)];
}

// MODE 0: LN(a+b)   MODE 1: LN(ga*a + gb*b), gates pre-sigmoided
// MODE 2: LN(clip(a+b, +-65504))
template <int MODE>
__global__ void __launch_bounds__(256)
fuse_ln(const float* __restrict__ a, const float* __restrict__ b,
        const float* __restrict__ gates, const float* __restrict__ gamma,
        const float* __restrict__ beta, float* __restrict__ out) {
  __shared__ float red[256];
  const int row = blockIdx.x, t = threadIdx.x;
  const size_t o = (size_t)row * 256 + t;
  float v;
  if (MODE == 1) {
    const float ga = gates[(size_t)row * 512 + t];
    const float gb = gates[(size_t)row * 512 + 256 + t];
    v = ga * a[o] + gb * b[o];
  } else {
    v = a[o] + b[o];
    if (MODE == 2) v = fminf(fmaxf(v, -65504.f), 65504.f);
  }
  red[t] = v; __syncthreads();
  for (int s = 128; s > 0; s >>= 1) { if (t < s) red[t] += red[t + s]; __syncthreads(); }
  const float mean = red[0] * (1.f / 256.f);
  __syncthreads();
  const float diff = v - mean;
  red[t] = diff * diff; __syncthreads();
  for (int s = 128; s > 0; s >>= 1) { if (t < s) red[t] += red[t + s]; __syncthreads(); }
  const float var = red[0] * (1.f / 256.f);
  out[o] = diff * rsqrtf(var + 1e-5f) * gamma[t] + beta[t];
}

__global__ void aw_softmax(const float* __restrict__ in, float* __restrict__ out,
                           int rows) {
  const int r = blockIdx.x * 256 + threadIdx.x;
  if (r >= rows) return;
  const float* p = in + (size_t)r * 16;
  float m = -1e30f;
#pragma unroll
  for (int i = 0; i < 16; ++i) m = fmaxf(m, p[i]);
  float e[16]; float s = 0.f;
#pragma unroll
  for (int i = 0; i < 16; ++i) { e[i] = __expf(p[i] - m); s += e[i]; }
  const float inv = 1.f / s;
  float* q = out + (size_t)r * 16;
#pragma unroll
  for (int i = 0; i < 16; ++i) q[i] = e[i] * inv;
}

// ---------------------------------------------------------------------------
// MS-deformable attention sampling: block = (b,q) query, thread = (head, dh).
// ---------------------------------------------------------------------------
__device__ __forceinline__ float dsamp(const float* __restrict__ base, int y,
                                       int x, int H, int W, int c) {
  if (x < 0 || x >= W || y < 0 || y >= H) return 0.f;
  return base[((size_t)y * W + x) * 256 + c];
}

__global__ void __launch_bounds__(256)
deform_attn(const float* __restrict__ value, const float* __restrict__ so,
            const float* __restrict__ aw, const float* __restrict__ ref,
            float* __restrict__ out) {
  const int bq = blockIdx.x;              // 0..9599
  const int b = bq / 300;
  const int h = threadIdx.x >> 5, d = threadIdx.x & 31;
  const float* rp = ref + (size_t)bq * 4;
  const float rx = rp[0], ry = rp[1], rw = rp[2], rh = rp[3];
  const float* sop = so + (size_t)bq * 256 + h * 32;  // [p*2 + xy]
  const float* awp = aw + (size_t)bq * 128 + h * 16;
  const float* vb = value + (size_t)b * 8500 * 256;
  const int   dims[4]   = {80, 40, 20, 10};
  const int   starts[4] = {0, 6400, 8000, 8400};
  const int   c = h * 32 + d;
  float acc = 0.f;
#pragma unroll
  for (int p = 0; p < 16; ++p) {
    const int lvl = p >> 2;
    const int HH = dims[lvl], WW = dims[lvl];
    const float a  = awp[p];
    const float lx = rx + sop[p * 2 + 0] * 0.125f * rw;  // nps(0.25)*scale(0.5)
    const float ly = ry + sop[p * 2 + 1] * 0.125f * rh;
    const float x = lx * WW - 0.5f;
    const float y = ly * HH - 0.5f;
    const float xf = floorf(x), yf = floorf(y);
    const int x0 = (int)xf, y0 = (int)yf;
    const float dx = x - xf, dy = y - yf;
    const float* base = vb + (size_t)starts[lvl] * 256;
    const float v00 = dsamp(base, y0,     x0,     HH, WW, c);
    const float v01 = dsamp(base, y0,     x0 + 1, HH, WW, c);
    const float v10 = dsamp(base, y0 + 1, x0,     HH, WW, c);
    const float v11 = dsamp(base, y0 + 1, x0 + 1, HH, WW, c);
    acc += a * (v00 * (1.f - dy) * (1.f - dx) + v01 * (1.f - dy) * dx +
                v10 * dy * (1.f - dx)         + v11 * dy * dx);
  }
  out[(size_t)bq * 256 + c] = acc;
}

// ---------------------------------------------------------------------------
// Launch sequence
// ---------------------------------------------------------------------------
extern "C" void kernel_launch(void* const* d_in, const int* in_sizes, int n_in,
                              void* d_out, int out_size, void* d_ws, size_t ws_size,
                              hipStream_t stream) {
  (void)in_sizes; (void)n_in; (void)out_size; (void)ws_size;
  const float* target = (const float*)d_in[0];
  const float* refpts = (const float*)d_in[1];
  const float* value  = (const float*)d_in[2];
  const float* pos    = (const float*)d_in[3];
  const float* Wq  = (const float*)d_in[4];  const float* bq  = (const float*)d_in[5];
  const float* Wk  = (const float*)d_in[6];  const float* bk  = (const float*)d_in[7];
  const float* Wv  = (const float*)d_in[8];  const float* bv  = (const float*)d_in[9];
  const float* Wo  = (const float*)d_in[10]; const float* bo  = (const float*)d_in[11];
  const float* g1  = (const float*)d_in[12]; const float* be1 = (const float*)d_in[13];
  const float* soW = (const float*)d_in[14]; const float* sob = (const float*)d_in[15];
  const float* awW = (const float*)d_in[16]; const float* awb = (const float*)d_in[17];
  const float* gW  = (const float*)d_in[18]; const float* gbv = (const float*)d_in[19];
  const float* g2  = (const float*)d_in[20]; const float* be2 = (const float*)d_in[21];
  const float* l1W = (const float*)d_in[22]; const float* l1b = (const float*)d_in[23];
  const float* l2W = (const float*)d_in[24]; const float* l2b = (const float*)d_in[25];
  const float* g3  = (const float*)d_in[26]; const float* be3 = (const float*)d_in[27];

  float* ws = (float*)d_ws;
  constexpr size_t SZ = 9600u * 256u;   // 2,457,600 floats
  float* qin   = ws + 0 * SZ;
  float* Qb    = ws + 1 * SZ;
  float* Kb    = ws + 2 * SZ;
  float* Vb    = ws + 3 * SZ;
  float* attn  = ws + 0 * SZ;           // reuse qin
  float* proj  = ws + 1 * SZ;           // reuse Q
  float* t     = ws + 2 * SZ;           // reuse K
  float* qc    = ws + 3 * SZ;           // reuse V
  float* so    = ws + 0 * SZ;           // reuse attn
  float* awr   = ws + 1 * SZ;           // reuse proj
  float* aw    = ws + 1 * SZ + 9600u * 128u;
  float* t2    = ws + 4 * SZ;
  float* cat   = ws + 5 * SZ;           // 2*SZ
  float* gates = ws + 0 * SZ;           // 2*SZ (so/awr dead)
  float* tg    = ws + 3 * SZ;           // reuse qc
  float* ffn1  = ws + 7 * SZ;           // 4*SZ (9600x1024)
  float* ffn2  = ws + 0 * SZ;           // gates dead
  // total workspace: 11*SZ floats = ~108.1 MB

  const int M = 9600;                   // 75 tiles of 128 rows
  const dim3 B128(128), B256(256);

  // ---- self-attention
  add2<<<dim3(SZ / 256), B256, 0, stream>>>(target, pos, qin, (int)SZ);
  gemm_wmma<ACT_NONE><<<dim3(4, 75),  B128, 0, stream>>>(qin,    Wq, bq, Qb,   M, 256, 256);
  gemm_wmma<ACT_NONE><<<dim3(4, 75),  B128, 0, stream>>>(qin,    Wk, bk, Kb,   M, 256, 256);
  gemm_wmma<ACT_NONE><<<dim3(4, 75),  B128, 0, stream>>>(target, Wv, bv, Vb,   M, 256, 256);
  attn_flash<<<dim3(5, 8, 32), B128, 0, stream>>>(Qb, Kb, Vb, attn);
  gemm_wmma<ACT_NONE><<<dim3(4, 75),  B128, 0, stream>>>(attn,   Wo, bo, proj, M, 256, 256);
  fuse_ln<0><<<dim3(M), B256, 0, stream>>>(target, proj, nullptr, g1, be1, t);

  // ---- deformable cross-attention
  add2<<<dim3(SZ / 256), B256, 0, stream>>>(t, pos, qc, (int)SZ);
  gemm_wmma<ACT_NONE><<<dim3(4, 75), B128, 0, stream>>>(qc, soW, sob, so,  M, 256, 256);
  gemm_wmma<ACT_NONE><<<dim3(2, 75), B128, 0, stream>>>(qc, awW, awb, awr, M, 128, 256);
  aw_softmax<<<dim3(300), B256, 0, stream>>>(awr, aw, 76800);
  deform_attn<<<dim3(M), B256, 0, stream>>>(value, so, aw, refpts, t2);

  // ---- gated fusion
  concat2<<<dim3(2 * SZ / 256), B256, 0, stream>>>(t, t2, cat, (int)(2 * SZ));
  gemm_wmma<ACT_SIG><<<dim3(8, 75), B128, 0, stream>>>(cat, gW, gbv, gates, M, 512, 512);
  fuse_ln<1><<<dim3(M), B256, 0, stream>>>(t, t2, gates, g2, be2, tg);

  // ---- FFN
  gemm_wmma<ACT_RELU><<<dim3(16, 75), B128, 0, stream>>>(tg,   l1W, l1b, ffn1, M, 1024, 256);
  gemm_wmma<ACT_NONE><<<dim3(4, 75),  B128, 0, stream>>>(ffn1, l2W, l2b, ffn2, M, 256, 1024);
  fuse_ln<2><<<dim3(M), B256, 0, stream>>>(tg, ffn2, nullptr, g3, be3, (float*)d_out);
}
